// MultiHeadedAttention_1288490189540
// MI455X (gfx1250) — compile-verified, matrix operand order checked by reading, not run on hardware
//
#include <hip/hip_runtime.h>
#include <hip/hip_bf16.h>
#include <stdint.h>

// Problem constants (reference: B=4, S=2048, D=1024, H=16, HD=64)
#define BQ 4
#define SS 2048
#define DD 1024
#define HH 16
#define HD 64
#define MM (BQ * SS)   // 8192 rows in the flattened (B*S, D) GEMMs

typedef __attribute__((ext_vector_type(16))) __bf16 v16bf;
typedef __attribute__((ext_vector_type(8)))  __bf16 v8bf;
typedef __attribute__((ext_vector_type(8)))  float  v8f;

// D = A(16x32 bf16) * B(32x16 bf16) + C(16x16 f32), wave32 WMMA.
static __device__ __forceinline__ v8f wmma_bf16(v16bf a, v16bf b, v8f c) {
  return __builtin_amdgcn_wmma_f32_16x16x32_bf16(false, a, false, b,
                                                 (short)0, c, false, false);
}

// Async DMA: copy 16 bytes from global memory directly into LDS (no VGPR
// data path, tracked by ASYNCcnt). lds_off is the byte offset within the
// workgroup's LDS allocation (low 32 bits of a flat shared pointer).
static __device__ __forceinline__ void async_copy16(uint32_t lds_off,
                                                    const void* gptr) {
  asm volatile("global_load_async_to_lds_b128 %0, %1, off"
               :: "v"(lds_off), "v"((uint64_t)(uintptr_t)gptr)
               : "memory");
}
static __device__ __forceinline__ void wait_async0() {
  asm volatile("s_wait_asynccnt 0x0" ::: "memory");
}

// Load one 16x32 bf16 WMMA operand fragment from a row-major array with K
// contiguous (works for A: rows = M, and for B: rows = N, since B[k][n] reads
// row n of the source with K contiguous).
// Per-lane layout (ISA 7.12.2, 16-bit A 16x32): lane = hf*16 + m,
//   el[0..7]  = src[m][hf*8 + 0..7]
//   el[8..15] = src[m][16 + hf*8 + 0..7]
// stride is in elements; stride*2 bytes must be a multiple of 16.
static __device__ __forceinline__ v16bf load_frag(const __bf16* base, int stride) {
  const int lane = threadIdx.x & 31;
  const int m    = lane & 15;
  const int hf   = lane >> 4;
  const __bf16* p = base + (size_t)m * stride + hf * 8;
  v16bf f;
  *(v8bf*)&f         = *(const v8bf*)p;
  *(((v8bf*)&f) + 1) = *(const v8bf*)(p + 16);
  return f;
}

// ---------------------------------------------------------------------------
// One-shot fp32 -> bf16 conversion (for weight matrices), 4 elems/thread.
// ---------------------------------------------------------------------------
__global__ __launch_bounds__(256)
void f32_to_bf16_kernel(const float* __restrict__ in, void* __restrict__ outv,
                        int n) {
  __bf16* out = (__bf16*)outv;
  const int i = (blockIdx.x * 256 + threadIdx.x) * 4;
  if (i + 3 < n) {
    const float4 v = *(const float4*)(in + i);
    out[i + 0] = (__bf16)v.x;
    out[i + 1] = (__bf16)v.y;
    out[i + 2] = (__bf16)v.z;
    out[i + 3] = (__bf16)v.w;
  }
}

// ---------------------------------------------------------------------------
// Tiled GEMM: Y = X @ W^T + bias   (X: MMxDD, W: DDxDD bf16 pre-converted)
// Block tile 64x64, K step 32, 4 waves (128 threads).
// W tile is staged via async global->LDS DMA (ASYNCcnt). X tile is staged
// async too when already bf16 (IN_IS_BF16), else converted via VALU.
// MODE 0: write bf16 (b,h,s,hd)      (Q, K projections)
// MODE 1: write bf16 (b,h,hd,s)      (V projection, pre-transposed for PV)
// MODE 2: write fp32 row-major (m,n) (final output projection)
// ---------------------------------------------------------------------------
#define BM 64
#define BN 64
#define BK 32
#define PSTR 40   // LDS row stride (elems): 40*2B = 80B, multiple of 16B

template<int IN_IS_BF16, int MODE>
__global__ __launch_bounds__(128)
void proj_gemm_kernel(const void* __restrict__ Xv, const void* __restrict__ Wbv,
                      const float* __restrict__ bias, void* __restrict__ outv) {
  __shared__ __bf16 Xs[BM * PSTR];
  __shared__ __bf16 Ws[BM * PSTR];
  const int tid  = threadIdx.x;
  const int wave = tid >> 5;
  const int lane = tid & 31;
  const int hlf  = lane >> 4;
  const int ln   = lane & 15;
  const int m0 = blockIdx.x * BM;
  const int n0 = blockIdx.y * BN;

  const float*  Xf = (const float*)Xv;
  const __bf16* Xb = (const __bf16*)Xv;
  const __bf16* Wb = (const __bf16*)Wbv;
  const uint32_t xs_base = (uint32_t)(uintptr_t)&Xs[0];
  const uint32_t ws_base = (uint32_t)(uintptr_t)&Ws[0];

  v8f acc[4];
#pragma unroll
  for (int c = 0; c < 4; ++c)
#pragma unroll
    for (int r = 0; r < 8; ++r) acc[c][r] = 0.0f;

  for (int kk = 0; kk < DD; kk += BK) {
    // --- W tile: 64 rows x 32 bf16 cols = 256 x 16B chunks, async DMA ---
#pragma unroll
    for (int i = 0; i < 2; ++i) {
      const int c   = tid + i * 128;
      const int row = c >> 2;          // 4 chunks per row
      const int q   = c & 3;
      async_copy16(ws_base + (uint32_t)(row * (PSTR * 2) + q * 16),
                   (const char*)Wb + ((size_t)(n0 + row) * DD + kk + q * 8) * 2);
      if (IN_IS_BF16) {
        async_copy16(xs_base + (uint32_t)(row * (PSTR * 2) + q * 16),
                     (const char*)Xb + ((size_t)(m0 + row) * DD + kk + q * 8) * 2);
      }
    }
    if (!IN_IS_BF16) {
      // fp32 X: load, convert to bf16, store to LDS (16 elems/thread).
#pragma unroll
      for (int i = 0; i < 16; ++i) {
        const int e   = tid + i * 128;
        const int row = e >> 5;
        const int col = e & 31;
        Xs[row * PSTR + col] =
            (__bf16)Xf[(size_t)(m0 + row) * DD + kk + col];
      }
      if (kk + BK < DD)   // prefetch next fp32 X tile (global_prefetch_b8)
        __builtin_prefetch(&Xf[(size_t)(m0 + (tid & 63)) * DD + kk + BK], 0, 0);
    }
    wait_async0();        // ASYNCcnt is not covered by __syncthreads()
    __syncthreads();

    const v16bf a = load_frag(&Xs[(wave * 16) * PSTR], PSTR);
#pragma unroll
    for (int cg = 0; cg < 4; ++cg) {
      const v16bf bfrag = load_frag(&Ws[(cg * 16) * PSTR], PSTR);
      acc[cg] = wmma_bf16(a, bfrag, acc[cg]);
    }
    __syncthreads();
  }

  // Epilogue: bias add + layout-specific store.
  // Accumulator layout: el[r] -> row (r + 8*hlf), col ln (per 16-col group).
#pragma unroll
  for (int cg = 0; cg < 4; ++cg) {
    const int n  = n0 + cg * 16 + ln;
    const float bv = bias[n];
#pragma unroll
    for (int r = 0; r < 8; ++r) {
      const int m = m0 + wave * 16 + r + 8 * hlf;
      const float y = acc[cg][r] + bv;
      if (MODE == 2) {
        ((float*)outv)[(size_t)m * DD + n] = y;
      } else {
        const int b = m >> 11, s = m & (SS - 1);   // S = 2048
        const int h = n >> 6,  hd = n & 63;        // HD = 64
        __bf16* ob = (__bf16*)outv;
        if (MODE == 0)
          ob[((size_t)(b * HH + h) * SS + s) * HD + hd] = (__bf16)y;
        else
          ob[((size_t)(b * HH + h) * HD + hd) * SS + s] = (__bf16)y;
      }
    }
  }
}

// ---------------------------------------------------------------------------
// Flash attention: one wave per 16-query tile, 4 waves/block.
// Q,K bf16 in (b,h,s,hd); V bf16 pre-transposed (b,h,hd,s); online softmax.
// ---------------------------------------------------------------------------
__global__ __launch_bounds__(128)
void flash_attn_kernel(const void* __restrict__ qv, const void* __restrict__ kv,
                       const void* __restrict__ vtv,
                       const uint8_t* __restrict__ mask,
                       void* __restrict__ attv) {
  __shared__ __bf16 Pst[4][16 * 32];     // per-wave P staging (acc->A layout)
  const int tid  = threadIdx.x;
  const int wave = tid >> 5;
  const int lane = tid & 31;
  const int hlf  = lane >> 4;
  const int ln   = lane & 15;
  const int bh = blockIdx.y;
  const int b  = bh >> 4;                // H = 16
  const int h  = bh & 15;
  const int q0 = (blockIdx.x * 4 + wave) * 16;
  const float NEGINF = -__builtin_inff();

  const __bf16* qb  = (const __bf16*)qv  + ((size_t)bh * SS + q0) * HD;
  const __bf16* kb  = (const __bf16*)kv  + (size_t)bh * SS * HD;
  const __bf16* vtb = (const __bf16*)vtv + (size_t)bh * HD * SS;

  // Q fragments: hd 0..31 and 32..63 (K dimension of QK^T).
  const v16bf aq0 = load_frag(qb, HD);
  const v16bf aq1 = load_frag(qb + 32, HD);

  float mi[8], li[8];
  v8f o[4];
#pragma unroll
  for (int r = 0; r < 8; ++r) { mi[r] = NEGINF; li[r] = 0.0f; }
#pragma unroll
  for (int cg = 0; cg < 4; ++cg)
#pragma unroll
    for (int r = 0; r < 8; ++r) o[cg][r] = 0.0f;

  for (int k0 = 0; k0 < SS; k0 += 32) {
    // Scores S = Q @ K^T / 8 for 32 keys (two 16x16 tiles).
    v8f sc[2];
#pragma unroll
    for (int t = 0; t < 2; ++t) {
      const v16bf bk0 = load_frag(kb + (size_t)(k0 + t * 16) * HD, HD);
      const v16bf bk1 = load_frag(kb + (size_t)(k0 + t * 16) * HD + 32, HD);
      v8f z;
#pragma unroll
      for (int r = 0; r < 8; ++r) z[r] = 0.0f;
      z = wmma_bf16(aq0, bk0, z);
      z = wmma_bf16(aq1, bk1, z);
      sc[t] = z;
    }
    // Scale (1/sqrt(64)) and apply boolean mask (-inf where masked).
#pragma unroll
    for (int t = 0; t < 2; ++t)
#pragma unroll
      for (int r = 0; r < 8; ++r) {
        const int qr = q0 + r + 8 * hlf;
        const int kc = k0 + t * 16 + ln;
        const float v = sc[t][r] * 0.125f;
        sc[t][r] = mask[(size_t)(b * SS + qr) * SS + kc] ? NEGINF : v;
      }
    // Row max across the 16 lanes of each half-wave.
    float rmax[8];
#pragma unroll
    for (int r = 0; r < 8; ++r) rmax[r] = fmaxf(sc[0][r], sc[1][r]);
#pragma unroll
    for (int d = 1; d < 16; d <<= 1)
#pragma unroll
      for (int r = 0; r < 8; ++r)
        rmax[r] = fmaxf(rmax[r], __shfl_xor(rmax[r], d, 32));

    float mn[8], al[8], rs[8];
#pragma unroll
    for (int r = 0; r < 8; ++r) {
      mn[r] = fmaxf(mi[r], rmax[r]);
      al[r] = (mi[r] == NEGINF) ? 0.0f : __expf(mi[r] - mn[r]);
      const bool live = (mn[r] != NEGINF);
      const float p0 = live ? __expf(sc[0][r] - mn[r]) : 0.0f;
      const float p1 = live ? __expf(sc[1][r] - mn[r]) : 0.0f;
      sc[0][r] = p0; sc[1][r] = p1;
      rs[r] = p0 + p1;
    }
#pragma unroll
    for (int d = 1; d < 16; d <<= 1)
#pragma unroll
      for (int r = 0; r < 8; ++r) rs[r] += __shfl_xor(rs[r], d, 32);
#pragma unroll
    for (int r = 0; r < 8; ++r) { li[r] = li[r] * al[r] + rs[r]; mi[r] = mn[r]; }
#pragma unroll
    for (int cg = 0; cg < 4; ++cg)
#pragma unroll
      for (int r = 0; r < 8; ++r) o[cg][r] *= al[r];

    // Stage P (accumulator layout -> row-major [16 q][32 k] bf16 in LDS),
    // then reload as the A fragment of P @ V. Single-wave RAW; the compiler
    // inserts the required s_wait_dscnt.
    __bf16* pw = &Pst[wave][0];
#pragma unroll
    for (int t = 0; t < 2; ++t)
#pragma unroll
      for (int r = 0; r < 8; ++r)
        pw[(r + 8 * hlf) * 32 + t * 16 + ln] = (__bf16)sc[t][r];
    const v16bf ap = load_frag(pw, 32);
#pragma unroll
    for (int cg = 0; cg < 4; ++cg) {
      const v16bf bv = load_frag(vtb + (size_t)(cg * 16) * SS + k0, SS);
      o[cg] = wmma_bf16(ap, bv, o[cg]);
    }
  }

  // Normalize and write attention output as bf16 (B, S, D) for the final GEMM.
  __bf16* att = (__bf16*)attv;
#pragma unroll
  for (int r = 0; r < 8; ++r) {
    const float inv = 1.0f / li[r];
    const int qr = q0 + r + 8 * hlf;
#pragma unroll
    for (int cg = 0; cg < 4; ++cg) {
      const int n = h * HD + cg * 16 + ln;
      att[(size_t)(b * SS + qr) * DD + n] = (__bf16)(o[cg][r] * inv);
    }
  }
}

// ---------------------------------------------------------------------------
extern "C" void kernel_launch(void* const* d_in, const int* in_sizes, int n_in,
                              void* d_out, int out_size, void* d_ws, size_t ws_size,
                              hipStream_t stream) {
  (void)in_sizes; (void)n_in; (void)out_size; (void)ws_size;
  const void*    xq   = d_in[0];
  const void*    xk   = d_in[1];
  const void*    xv   = d_in[2];
  const uint8_t* mask = (const uint8_t*)d_in[3];
  const float*   Wq = (const float*)d_in[4];
  const float*   bq = (const float*)d_in[5];
  const float*   Wk = (const float*)d_in[6];
  const float*   bk = (const float*)d_in[7];
  const float*   Wv = (const float*)d_in[8];
  const float*   bv = (const float*)d_in[9];
  const float*   Wo = (const float*)d_in[10];
  const float*   bo = (const float*)d_in[11];

  // Workspace layout (bytes):
  //   4 x MM*DD bf16 activations (16 MB each): q, k, vT, attn
  //   4 x DD*DD bf16 weights (2 MB each): wq, wk, wv, wo      => 72 MB total
  const size_t NMB = (size_t)MM * DD * 2;
  const size_t NWB = (size_t)DD * DD * 2;
  char* ws = (char*)d_ws;
  void* qbuf  = ws;
  void* kbuf  = ws + NMB;
  void* vtbuf = ws + 2 * NMB;
  void* abuf  = ws + 3 * NMB;
  void* wqb   = ws + 4 * NMB;
  void* wkb   = ws + 4 * NMB + NWB;
  void* wvb   = ws + 4 * NMB + 2 * NWB;
  void* wob   = ws + 4 * NMB + 3 * NWB;

  const int nW = DD * DD;
  const dim3 cgrid(nW / (256 * 4)), cblk(256);
  f32_to_bf16_kernel<<<cgrid, cblk, 0, stream>>>(Wq, wqb, nW);
  f32_to_bf16_kernel<<<cgrid, cblk, 0, stream>>>(Wk, wkb, nW);
  f32_to_bf16_kernel<<<cgrid, cblk, 0, stream>>>(Wv, wvb, nW);
  f32_to_bf16_kernel<<<cgrid, cblk, 0, stream>>>(Wo, wob, nW);

  dim3 gg(MM / BM, DD / BN);   // 128 x 16 blocks
  dim3 bg(128);
  proj_gemm_kernel<0, 0><<<gg, bg, 0, stream>>>(xq, wqb, bq, qbuf);
  proj_gemm_kernel<0, 0><<<gg, bg, 0, stream>>>(xk, wkb, bk, kbuf);
  proj_gemm_kernel<0, 1><<<gg, bg, 0, stream>>>(xv, wvb, bv, vtbuf);
  flash_attn_kernel<<<dim3(SS / 64, BQ * HH), bg, 0, stream>>>(
      qbuf, kbuf, vtbuf, mask, abuf);
  proj_gemm_kernel<1, 2><<<gg, bg, 0, stream>>>(abuf, wob, bo, d_out);
}
